// FEPModel_29875792511791
// MI455X (gfx1250) — matrix operand order repaired and checked by reading, not compiled, not run on hardware
//
#include <hip/hip_runtime.h>
#include <cmath>

// ---------------- types ----------------
typedef __attribute__((ext_vector_type(16))) __bf16 v16bf;
typedef __attribute__((ext_vector_type(8)))  float  v8f;
typedef __attribute__((ext_vector_type(4)))  unsigned int v4u;
typedef __attribute__((ext_vector_type(4)))  int    v4i;
typedef __attribute__((ext_vector_type(8)))  int    v8i;

#define Bsz 4
#define Ssz 2048
#define Dsz 1024
#define NHh 4
#define DHh 256
#define CSz 256
#define NCc 8
#define Mrows 8192
#define BKk 64

#if __has_builtin(__builtin_amdgcn_tensor_load_to_lds) && __has_builtin(__builtin_amdgcn_s_wait_tensorcnt)
#define HAS_TDM 1
#endif

// ================= fp32 -> bf16 with transpose =================
// src: [K][N] f32  ->  dst: [N][K] bf16
__global__ __launch_bounds__(256) void f2bf_t_kernel(const float* __restrict__ s,
                                                     __bf16* __restrict__ d,
                                                     int K, int N) {
    __shared__ float tile[64][65];
    int k0 = blockIdx.y * 64, n0 = blockIdx.x * 64;
    for (int idx = threadIdx.x; idx < 64 * 64; idx += 256) {
        int r = idx >> 6, c = idx & 63;
        tile[r][c] = s[(size_t)(k0 + r) * N + n0 + c];
    }
    __syncthreads();
    for (int idx = threadIdx.x; idx < 64 * 64; idx += 256) {
        int r = idx >> 6, c = idx & 63;
        d[(size_t)(n0 + r) * K + k0 + c] = (__bf16)tile[c][r];
    }
}

// ================= embedding =================
__global__ __launch_bounds__(256) void embed_kernel(const int* __restrict__ ids,
                                                    const float* __restrict__ emb,
                                                    const float* __restrict__ pos,
                                                    float* __restrict__ h) {
    int m = blockIdx.x;
    int s = m & (Ssz - 1);
    int id = ids[m];
    const float* e = emb + (size_t)id * Dsz;
    const float* p = pos + (size_t)s * Dsz;
    float* o = h + (size_t)m * Dsz;
    for (int d = threadIdx.x; d < Dsz; d += 256) o[d] = e[d] + p[d];
}

// ================= layernorm =================
__global__ __launch_bounds__(256) void ln_rows(const float* __restrict__ x,
                                               const float* __restrict__ g,
                                               const float* __restrict__ b,
                                               __bf16* __restrict__ obf,
                                               float* __restrict__ of32) {
    int m = blockIdx.x, t = threadIdx.x;
    const float* xr = x + (size_t)m * Dsz;
    __shared__ float red[256];
    float lv[4];
    float s = 0.f;
#pragma unroll
    for (int i = 0; i < 4; i++) { lv[i] = xr[i * 256 + t]; s += lv[i]; }
    red[t] = s; __syncthreads();
    for (int o = 128; o > 0; o >>= 1) { if (t < o) red[t] += red[t + o]; __syncthreads(); }
    float mu = red[0] * (1.f / Dsz);
    __syncthreads();
    float v = 0.f;
#pragma unroll
    for (int i = 0; i < 4; i++) { float dd = lv[i] - mu; v += dd * dd; }
    red[t] = v; __syncthreads();
    for (int o = 128; o > 0; o >>= 1) { if (t < o) red[t] += red[t + o]; __syncthreads(); }
    float rs = rsqrtf(red[0] * (1.f / Dsz) + 1e-5f);
#pragma unroll
    for (int i = 0; i < 4; i++) {
        int d = i * 256 + t;
        float y = (lv[i] - mu) * rs * g[d] + b[d];
        if (obf)  obf[(size_t)m * Dsz + d] = (__bf16)y;
        if (of32) of32[(size_t)m * Dsz + d] = y;
    }
}

// ================= TDM tile load (2D, bf16) =================
#ifdef HAS_TDM
__device__ __forceinline__ void tdm_load_tile(const __bf16* g, unsigned lds,
                                              int rowlen, int nrows) {
    unsigned long long ga = (unsigned long long)(size_t)g;
    v4u g0;
    g0.x = 1u;                                            // count=1
    g0.y = lds;                                           // lds_addr
    g0.z = (unsigned)ga;                                  // global_addr[31:0]
    g0.w = ((unsigned)(ga >> 32) & 0x01FFFFFFu) | 0x80000000u;  // addr[56:32] | type=2
    v8i g1;
    g1[0] = 1 << 16;                                      // data_size = 2B; no mask/flags
    g1[1] = (rowlen & 0xFFFF) << 16;                      // tensor_dim0 lo16
    g1[2] = ((rowlen >> 16) & 0xFFFF) | ((nrows & 0xFFFF) << 16);   // dim0 hi | dim1 lo
    g1[3] = ((nrows >> 16) & 0xFFFF) | (BKk << 16);       // dim1 hi | tile_dim0=64
    g1[4] = 128;                                          // tile_dim1=128, tile_dim2=0
    g1[5] = rowlen;                                       // tensor_dim0_stride lo32
    g1[6] = 0;
    g1[7] = 0;
    v4i z4 = {0, 0, 0, 0};
    v8i z8 = {0, 0, 0, 0, 0, 0, 0, 0};
    __builtin_amdgcn_tensor_load_to_lds(g0, g1, z4, z4, z8, 0);
}
#endif

// ================= WMMA bf16 GEMM =================
// C[M,N] = A[M,K](bf16) @ Bt[N,K](bf16, pre-transposed) + bias (+resid) (+gelu)
// Block tile 128x128, BK=64; 8 waves (4x2), each wave 2(M)x4(N) wmma tiles.
__global__ __launch_bounds__(256) void gemm_bf16(const __bf16* __restrict__ A,
                                                 const __bf16* __restrict__ Bt,
                                                 const float* __restrict__ bias,
                                                 const float* __restrict__ resid,
                                                 float* __restrict__ Cf,
                                                 __bf16* __restrict__ Cb,
                                                 int M, int N, int K, int gelu) {
    __shared__ __align__(16) __bf16 As[2][128 * BKk];
    __shared__ __align__(16) __bf16 Bs[2][128 * BKk];
    const int n0 = blockIdx.x * 128, m0 = blockIdx.y * 128;
    const int t = threadIdx.x;
    const int wave = t >> 5, lane = t & 31;
    const int wm = wave >> 1, wn = wave & 1;
    const int lx = lane & 15, kh = lane >> 4;

    const v8f vzero = {0.f, 0.f, 0.f, 0.f, 0.f, 0.f, 0.f, 0.f};
    v8f acc[2][4];
#pragma unroll
    for (int i = 0; i < 2; i++)
#pragma unroll
        for (int j = 0; j < 4; j++) acc[i][j] = vzero;

    auto compute_tile = [&](const __bf16* Asb, const __bf16* Bsb) {
#pragma unroll
        for (int kk = 0; kk < 2; kk++) {
            v16bf af[2], bf[4];
#pragma unroll
            for (int ti = 0; ti < 2; ti++) {
                int arow = wm * 32 + ti * 16 + lx;
                ((float4*)&af[ti])[0] = *(const float4*)(Asb + arow * BKk + kk * 32 + kh * 8);
                ((float4*)&af[ti])[1] = *(const float4*)(Asb + arow * BKk + kk * 32 + 16 + kh * 8);
            }
#pragma unroll
            for (int tj = 0; tj < 4; tj++) {
                int bcol = wn * 64 + tj * 16 + lx;
                ((float4*)&bf[tj])[0] = *(const float4*)(Bsb + bcol * BKk + kk * 32 + kh * 8);
                ((float4*)&bf[tj])[1] = *(const float4*)(Bsb + bcol * BKk + kk * 32 + 16 + kh * 8);
            }
#pragma unroll
            for (int ti = 0; ti < 2; ti++)
#pragma unroll
                for (int tj = 0; tj < 4; tj++)
                    acc[ti][tj] = __builtin_amdgcn_wmma_f32_16x16x32_bf16(
                        false, af[ti], false, bf[tj], (short)0, acc[ti][tj], false, false);
        }
    };

#ifdef HAS_TDM
    // ---- double-buffered TDM staging: DMA next tile while computing current ----
    const bool issuer = (t == 0);
    if (issuer) {
        tdm_load_tile(A + (size_t)m0 * K, (unsigned)(size_t)&As[0][0], K, M);
        tdm_load_tile(Bt + (size_t)n0 * K, (unsigned)(size_t)&Bs[0][0], K, N);
        __builtin_amdgcn_s_wait_tensorcnt(0);
    }
    __syncthreads();
    int cur = 0;
    for (int k0 = 0; k0 < K; k0 += BKk) {
        if (issuer && (k0 + BKk < K)) {
            tdm_load_tile(A + (size_t)m0 * K + k0 + BKk, (unsigned)(size_t)&As[cur ^ 1][0], K, M);
            tdm_load_tile(Bt + (size_t)n0 * K + k0 + BKk, (unsigned)(size_t)&Bs[cur ^ 1][0], K, N);
        }
        compute_tile(&As[cur][0], &Bs[cur][0]);
        if (issuer) __builtin_amdgcn_s_wait_tensorcnt(0);
        __syncthreads();
        cur ^= 1;
    }
#else
    // ---- manual staging fallback (contiguous 128b loads/stores; Bt pre-transposed) ----
    const int ra = t >> 1, ca = (t & 1) * 32;
    for (int k0 = 0; k0 < K; k0 += BKk) {
        const __bf16* ag = A + (size_t)(m0 + ra) * K + k0 + ca;
        const __bf16* bg = Bt + (size_t)(n0 + ra) * K + k0 + ca;
        float4 av[4], bv[4];
#pragma unroll
        for (int i = 0; i < 4; i++) { av[i] = ((const float4*)ag)[i]; bv[i] = ((const float4*)bg)[i]; }
#pragma unroll
        for (int i = 0; i < 4; i++) {
            *(float4*)(&As[0][ra * BKk + ca] + i * 8) = av[i];
            *(float4*)(&Bs[0][ra * BKk + ca] + i * 8) = bv[i];
        }
        if (k0 + BKk < K) {
            __builtin_prefetch(ag + BKk, 0, 1);
            __builtin_prefetch(bg + BKk, 0, 1);
        }
        __syncthreads();
        compute_tile(&As[0][0], &Bs[0][0]);
        __syncthreads();
    }
#endif

    // ---- epilogue ----
#pragma unroll
    for (int ti = 0; ti < 2; ti++) {
#pragma unroll
        for (int tj = 0; tj < 4; tj++) {
            int col  = n0 + wn * 64 + tj * 16 + lx;
            int rowb = m0 + wm * 32 + ti * 16 + kh * 8;
            float bi = bias ? bias[col] : 0.f;
#pragma unroll
            for (int r = 0; r < 8; r++) {
                float v = acc[ti][tj][r] + bi;
                size_t idx = (size_t)(rowb + r) * N + col;
                if (resid) v += resid[idx];
                if (gelu)  v = 0.5f * v * (1.f + erff(v * 0.70710678118f));
                if (Cf) Cf[idx] = v;
                if (Cb) Cb[idx] = (__bf16)v;
            }
        }
    }
}

// ================= windowed attention =================
__global__ __launch_bounds__(256) void attn_win(const float* __restrict__ qkv,
                                                __bf16* __restrict__ outbf) {
    const int qt = blockIdx.x, bh = blockIdx.y;
    const int b = bh >> 2, h = bh & 3;
    const int q0 = qt * 64;
    const int t = threadIdx.x;
    __shared__ float Sc[64][128];

    for (int idx = t; idx < 64 * 128; idx += 256) {
        int q = idx >> 7, w = idx & 127;
        int i = q0 + q, j = i - 127 + w;
        float s = -__builtin_inff();
        if (j >= 0) {
            const float* qp = qkv + (size_t)(b * Ssz + i) * 3072 + h * DHh;
            const float* kp = qkv + (size_t)(b * Ssz + j) * 3072 + 1024 + h * DHh;
            float acc = 0.f;
            for (int d = 0; d < DHh; d++) acc += qp[d] * kp[d];
            s = acc * 0.0625f;
        }
        Sc[q][w] = s;
    }
    __syncthreads();
    if (t < 64) {
        float mx = -__builtin_inff();
        for (int w = 0; w < 128; w++) mx = fmaxf(mx, Sc[t][w]);
        float sm = 0.f;
        for (int w = 0; w < 128; w++) { float e = expf(Sc[t][w] - mx); Sc[t][w] = e; sm += e; }
        float inv = 1.f / sm;
        for (int w = 0; w < 128; w++) Sc[t][w] *= inv;
    }
    __syncthreads();
    for (int q = 0; q < 64; q++) {
        int i = q0 + q;
        int wlo = (i < 127) ? (127 - i) : 0;
        float acc = 0.f;
        for (int w = wlo; w < 128; w++) {
            int j = i - 127 + w;
            acc += Sc[q][w] * qkv[(size_t)(b * Ssz + j) * 3072 + 2048 + h * DHh + t];
        }
        outbf[(size_t)(b * Ssz + i) * Dsz + h * DHh + t] = (__bf16)acc;
    }
}

// ================= memory module =================
__global__ __launch_bounds__(256) void chunk_mean(const float* __restrict__ hn,
                                                  float* __restrict__ cm) {
    int n = blockIdx.x;
    int b = n >> 3, c = n & 7;
    for (int d = threadIdx.x; d < Dsz; d += 256) {
        const float* p = hn + (size_t)(b * Ssz + c * CSz) * Dsz + d;
        float s = 0.f;
        for (int r = 0; r < CSz; r++) s += p[(size_t)r * Dsz];
        cm[n * Dsz + d] = s * (1.f / CSz);
    }
}

__global__ __launch_bounds__(256) void mem_head(const float* __restrict__ cm,
                                                const float* __restrict__ qw,
                                                const float* __restrict__ qb,
                                                const float* __restrict__ pw,
                                                const float* __restrict__ pb,
                                                const float* __restrict__ Sm,
                                                float* __restrict__ retb,
                                                float* __restrict__ Fb, int hh) {
    int n = blockIdx.x, t = threadIdx.x;
    __shared__ float xr[Dsz];
    __shared__ float ql[64];
    __shared__ float red[256];
    for (int d = t; d < Dsz; d += 256) xr[d] = cm[n * Dsz + d];
    __syncthreads();
    float qv = 0.f;
    if (t < 64) {
        float a = qb[t];
        for (int k = 0; k < Dsz; k++) a += xr[k] * qw[k * 64 + t];
        qv = a;
    }
    red[t] = (t < 64) ? qv * qv : 0.f; __syncthreads();
    for (int o = 128; o > 0; o >>= 1) { if (t < o) red[t] += red[t + o]; __syncthreads(); }
    float nrm = sqrtf(red[0]);
    if (t < 64) ql[t] = qv / fmaxf(nrm, 1e-12f);
    __syncthreads();

    int b = n >> 3, c = n & 7;
    int tn = b * NCc + ((c < NCc - 1) ? c + 1 : NCc - 1);
    float Fp = 0.f;
    for (int d = t; d < Dsz; d += 256) {
        float r = 0.f;
        for (int k = 0; k < 64; k++) r += ql[k] * Sm[k * Dsz + d];
        float a = pb[d];
        for (int k = 0; k < Dsz; k++) a += xr[k] * pw[k * Dsz + d];
        float sp = (a > 30.f) ? a : log1pf(expf(a));
        float prec = sp + 0.01f;
        float err = cm[tn * Dsz + d] - r;
        Fp += prec * err * err - logf(prec);
        retb[(size_t)(n * 3 + hh) * Dsz + d] = r;
    }
    red[t] = Fp; __syncthreads();
    for (int o = 128; o > 0; o >>= 1) { if (t < o) red[t] += red[t + o]; __syncthreads(); }
    if (t == 0) Fb[n * 3 + hh] = red[0] * (1.f / Dsz);
}

__global__ __launch_bounds__(256) void mem_combine(const float* __restrict__ Fb,
                                                   const float* __restrict__ retb,
                                                   float* __restrict__ mraw) {
    int n = blockIdx.x;
    __shared__ float wv[3];
    if (threadIdx.x == 0) {
        float f0 = -Fb[n * 3], f1 = -Fb[n * 3 + 1], f2 = -Fb[n * 3 + 2];
        float mx = fmaxf(f0, fmaxf(f1, f2));
        float e0 = expf(f0 - mx), e1 = expf(f1 - mx), e2 = expf(f2 - mx);
        float s = e0 + e1 + e2;
        wv[0] = e0 / s; wv[1] = e1 / s; wv[2] = e2 / s;
    }
    __syncthreads();
    for (int d = threadIdx.x; d < Dsz; d += 256)
        mraw[n * Dsz + d] = wv[0] * retb[(size_t)(n * 3) * Dsz + d]
                          + wv[1] * retb[(size_t)(n * 3 + 1) * Dsz + d]
                          + wv[2] * retb[(size_t)(n * 3 + 2) * Dsz + d];
}

__global__ __launch_bounds__(256) void mem_outp(const float* __restrict__ mraw,
                                                const float* __restrict__ ow,
                                                const float* __restrict__ ob,
                                                float* __restrict__ mout) {
    int n = blockIdx.x;
    __shared__ float xr[Dsz];
    for (int d = threadIdx.x; d < Dsz; d += 256) xr[d] = mraw[n * Dsz + d];
    __syncthreads();
    for (int d = threadIdx.x; d < Dsz; d += 256) {
        float a = ob[d];
        for (int k = 0; k < Dsz; k++) a += xr[k] * ow[k * Dsz + d];
        mout[n * Dsz + d] = a;
    }
}

__global__ __launch_bounds__(256) void mem_gate(const float* __restrict__ cm,
                                                const float* __restrict__ mout,
                                                const float* __restrict__ gw,
                                                const float* __restrict__ gb,
                                                float* __restrict__ mbl) {
    int n = blockIdx.x;
    __shared__ float xr[Dsz];
    __shared__ float mo[Dsz];
    for (int d = threadIdx.x; d < Dsz; d += 256) {
        xr[d] = cm[n * Dsz + d];
        mo[d] = mout[n * Dsz + d];
    }
    __syncthreads();
    for (int d = threadIdx.x; d < Dsz; d += 256) {
        float a = gb[d];
        for (int k = 0; k < Dsz; k++) a += xr[k] * gw[(size_t)k * Dsz + d];
        for (int k = 0; k < Dsz; k++) a += mo[k] * gw[(size_t)(k + Dsz) * Dsz + d];
        float g = 0.3f / (1.f + expf(-a));
        mbl[n * Dsz + d] = (1.f - g) * xr[d] + g * mo[d];
    }
}

__global__ __launch_bounds__(256) void mem_update(float* __restrict__ h,
                                                  const float* __restrict__ hn,
                                                  const float* __restrict__ mbl) {
    int m = blockIdx.x;
    int b = m >> 11, s = m & (Ssz - 1);
    int n = b * NCc + (s >> 8);
    for (int d = threadIdx.x; d < Dsz; d += 256)
        h[(size_t)m * Dsz + d] += mbl[n * Dsz + d] - hn[(size_t)m * Dsz + d];
}

__global__ void fe_kernel(const float* __restrict__ Fb, float* __restrict__ out) {
    __shared__ float r[32];
    int t = threadIdx.x;
    r[t] = fminf(Fb[t * 3], fminf(Fb[t * 3 + 1], Fb[t * 3 + 2]));
    __syncthreads();
    for (int o = 16; o > 0; o >>= 1) { if (t < o) r[t] += r[t + o]; __syncthreads(); }
    if (t == 0) out[0] = r[0] * (1.f / 32.f);
}

// ================= launch =================
extern "C" void kernel_launch(void* const* d_in, const int* in_sizes, int n_in,
                              void* d_out, int out_size, void* d_ws, size_t ws_size,
                              hipStream_t stream) {
    (void)in_sizes; (void)n_in; (void)out_size; (void)ws_size;
    // pytree (key-sorted) map: 0 byte_ids, 1 byte_emb, 2 head_b, 3 head_w, 4 lno_b, 5 lno_g,
    // 6..17 lower[0]{ff1_b,ff1_w,ff2_b,ff2_w,ln1_b,ln1_g,ln2_b,ln2_g,out_b,out_w,qkv_b,qkv_w},
    // 18..29 lower[1], 30 gate_b, 31 gate_w, 32..36 heads[0]{S,prec_b,prec_w,q_b,q_w},
    // 37..41 heads[1], 42..46 heads[2], 47 ln_b, 48 ln_g, 49 outp_b, 50 outp_w,
    // 51 pos_emb, 52..63 upper[0], 64..75 upper[1]
    const int* ids = (const int*)d_in[0];
    auto F = [&](int i) { return (const float*)d_in[i]; };

    char* W = (char*)d_ws;
    size_t off = 0;
    auto alloc = [&](size_t bytes) -> char* {
        char* p = W + off;
        off = (off + bytes + 255) & ~(size_t)255;
        return p;
    };
    const size_t QKV_W = (size_t)1024 * 3072, OUT_W = (size_t)1024 * 1024;
    const size_t FF1_W = (size_t)1024 * 3072, FF2_W = (size_t)3072 * 1024;
    const size_t BLK_W = QKV_W + OUT_W + FF1_W + FF2_W;
    __bf16* WBF   = (__bf16*)alloc((4 * BLK_W + (size_t)1024 * 256) * 2);
    float*  H     = (float*)alloc((size_t)Mrows * Dsz * 4);
    __bf16* LNBF  = (__bf16*)alloc((size_t)Mrows * Dsz * 2);
    float*  HN    = (float*)alloc((size_t)Mrows * Dsz * 4);
    char*   BIGR  = alloc((size_t)Mrows * 3072 * 4);
    float*  BIGF  = (float*)BIGR;
    __bf16* BIGBF = (__bf16*)BIGR;
    __bf16* ATTNBF= (__bf16*)alloc((size_t)Mrows * Dsz * 2);
    float*  CM    = (float*)alloc((size_t)32 * Dsz * 4);
    float*  RET   = (float*)alloc((size_t)32 * 3 * Dsz * 4);
    float*  FB    = (float*)alloc(32 * 3 * 4);
    float*  MRAW  = (float*)alloc((size_t)32 * Dsz * 4);
    float*  MOUT  = (float*)alloc((size_t)32 * Dsz * 4);
    float*  MBL   = (float*)alloc((size_t)32 * Dsz * 4);

    // convert + transpose W[K][N] -> Wt[N][K] bf16
    auto cvt_t = [&](const float* src, __bf16* dst, int K, int N) {
        f2bf_t_kernel<<<dim3(N / 64, K / 64), 256, 0, stream>>>(src, dst, K, N);
    };

    const int blockBase[4] = {6, 18, 52, 64};
    __bf16 *wqkv[4], *wout[4], *wff1[4], *wff2[4];
    for (int i = 0; i < 4; i++) {
        __bf16* base = WBF + (size_t)i * BLK_W;
        wqkv[i] = base;
        wout[i] = base + QKV_W;
        wff1[i] = base + QKV_W + OUT_W;
        wff2[i] = base + QKV_W + OUT_W + FF1_W;
        cvt_t(F(blockBase[i] + 11), wqkv[i], 1024, 3072);
        cvt_t(F(blockBase[i] + 9),  wout[i], 1024, 1024);
        cvt_t(F(blockBase[i] + 1),  wff1[i], 1024, 3072);
        cvt_t(F(blockBase[i] + 3),  wff2[i], 3072, 1024);
    }
    __bf16* WHEAD = WBF + (size_t)4 * BLK_W;
    cvt_t(F(3), WHEAD, 1024, 256);

    auto gemm = [&](const __bf16* A, const __bf16* Bt, const float* bias, const float* resid,
                    float* Cf, __bf16* Cb, int M, int N, int K, int gelu) {
        gemm_bf16<<<dim3(N / 128, M / 128), 256, 0, stream>>>(A, Bt, bias, resid, Cf, Cb, M, N, K, gelu);
    };

    embed_kernel<<<Mrows, 256, 0, stream>>>(ids, F(1), F(51), H);

    auto run_block = [&](int bi) {
        int bb = blockBase[bi];
        ln_rows<<<Mrows, 256, 0, stream>>>(H, F(bb + 5), F(bb + 4), LNBF, nullptr);
        gemm(LNBF, wqkv[bi], F(bb + 10), nullptr, BIGF, nullptr, Mrows, 3072, 1024, 0);
        attn_win<<<dim3(Ssz / 64, Bsz * NHh), 256, 0, stream>>>(BIGF, ATTNBF);
        gemm(ATTNBF, wout[bi], F(bb + 8), H, H, nullptr, Mrows, 1024, 1024, 0);
        ln_rows<<<Mrows, 256, 0, stream>>>(H, F(bb + 7), F(bb + 6), LNBF, nullptr);
        gemm(LNBF, wff1[bi], F(bb + 0), nullptr, nullptr, BIGBF, Mrows, 3072, 1024, 1);
        gemm(BIGBF, wff2[bi], F(bb + 2), H, H, nullptr, Mrows, 1024, 3072, 0);
    };

    run_block(0);
    run_block(1);

    ln_rows<<<Mrows, 256, 0, stream>>>(H, F(48), F(47), nullptr, HN);
    chunk_mean<<<32, 256, 0, stream>>>(HN, CM);
    for (int hh = 0; hh < 3; hh++) {
        int hb = 32 + hh * 5;
        mem_head<<<32, 256, 0, stream>>>(CM, F(hb + 4), F(hb + 3), F(hb + 2), F(hb + 1), F(hb + 0),
                                         RET, FB, hh);
    }
    mem_combine<<<32, 256, 0, stream>>>(FB, RET, MRAW);
    mem_outp<<<32, 256, 0, stream>>>(MRAW, F(50), F(49), MOUT);
    mem_gate<<<32, 256, 0, stream>>>(CM, MOUT, F(31), F(30), MBL);
    mem_update<<<Mrows, 256, 0, stream>>>(H, HN, MBL);

    run_block(2);
    run_block(3);

    ln_rows<<<Mrows, 256, 0, stream>>>(H, F(5), F(4), LNBF, nullptr);
    gemm(LNBF, WHEAD, F(2), nullptr, (float*)d_out, nullptr, Mrows, 256, 1024, 0);
    fe_kernel<<<1, 32, 0, stream>>>(FB, (float*)d_out + (size_t)Mrows * 256);
}
